// LigandDecoder_89799176225074
// MI455X (gfx1250) — matrix operand order repaired and checked by reading, not compiled, not run on hardware
//
#include <hip/hip_runtime.h>
#include <hip/hip_bf16.h>

typedef __attribute__((ext_vector_type(16))) _Float16 v16h;
typedef __attribute__((ext_vector_type(8)))  _Float16 v8h;
typedef __attribute__((ext_vector_type(8)))  float    v8f;

#define WMMA_F16(a,b,c) __builtin_amdgcn_wmma_f32_16x16x32_f16(false,(a),false,(b),(short)0,(c),false,false)

constexpr int BATCH = 16;
constexpr int NATOM = 128;
constexpr int BN    = BATCH * NATOM;

__device__ __forceinline__ float siluf(float x) { return x * (1.f / (1.f + __expf(-x))); }

// ---- WMMA fragment helpers (layouts per CDNA5 ISA §7.12.2) ----
// A matrix 16x32 f16: lanes 0-15 -> rows 0-15, halves 0..7 = K 0..7, 8..15 = K 16..23
//                     lanes 16-31 -> rows 0-15, halves 0..7 = K 8..15, 8..15 = K 24..31
__device__ __forceinline__ v16h a_frag(const _Float16* lds, int strideH, int kt, int lane) {
  int row  = lane & 15;
  int koff = kt * 32 + ((lane & 16) ? 8 : 0);
  const _Float16* base = lds + row * strideH + koff;
  v8h lo = *(const v8h*)base;
  v8h hi = *(const v8h*)(base + 16);
  return __builtin_shufflevector(lo, hi, 0,1,2,3,4,5,6,7,8,9,10,11,12,13,14,15);
}
// B matrix 32x16 f16, pre-packed so each lane holds its 16 halves contiguously.
__device__ __forceinline__ v16h b_frag(const _Float16* wpk, int f, int lane) {
  return ((const v16h*)wpk)[f * 32 + lane];
}

// Async copy of 32*16B per issue from global to LDS (ASYNCcnt path, no VGPR round trip).
// ldsOff/gaddr are per-lane. Caller must async_wait() before reading the LDS data.
__device__ __forceinline__ void async_b128(unsigned ldsOff, const void* gaddr) {
  asm volatile("global_load_async_to_lds_b128 %0, %1, off"
               :: "v"(ldsOff), "v"((unsigned long long)gaddr) : "memory");
}
__device__ __forceinline__ void async_wait() {
  asm volatile("s_wait_asynccnt 0x0" ::: "memory");
}
__device__ __forceinline__ unsigned lds_off(const void* p) {
  return (unsigned)(size_t)p;  // generic LDS pointer: low 32 bits are the LDS byte offset
}

// Pack f32 weight W[K][N] (row-major) into WMMA B-fragment order.
// frag f = kt*(N/16)+nt ; lane c<16: col = nt*16+c, K rows kt*32 + {0..7, 16..23}
//                         lane c>=16: same col, K rows kt*32 + {8..15, 24..31}
__global__ void pack_weight(const float* __restrict__ W, _Float16* __restrict__ dst, int K, int N) {
  int f = blockIdx.x, lane = threadIdx.x;
  int ntiles = N >> 4;
  int kt = f / ntiles, nt = f - kt * ntiles;
  int col   = nt * 16 + (lane & 15);
  int kbase = kt * 32 + ((lane & 16) ? 8 : 0);
  _Float16* o = dst + (size_t)f * 512 + lane * 16;
  for (int h = 0; h < 16; ++h) o[h] = (_Float16)W[(size_t)(kbase + h + (h & 8)) * N + col];
}

// ---- Generic WMMA GEMM: out[M,N] = (acc? out : 0) + A[M,K] @ W + bias, opt silu ----
// 1 wave per block; block tile = 16 rows x 64 cols. flags: 1=silu, 2=accumulate
__global__ __launch_bounds__(32) void gemm_wmma(const float* __restrict__ A,
                                                const _Float16* __restrict__ Wpk,
                                                const float* __restrict__ bias,
                                                float* __restrict__ out,
                                                int M, int K, int N, int flags) {
  extern __shared__ _Float16 ldsA[];  // 16*K halves
  int lane = threadIdx.x;
  int rowBase = blockIdx.x * 16;
  int colBase = blockIdx.y * 64;
  __builtin_prefetch(Wpk + ((size_t)(colBase >> 4)) * 512, 0, 0);
  for (int r = 0; r < 16; ++r)
    for (int k = lane; k < K; k += 32)
      ldsA[r * K + k] = (_Float16)A[(size_t)(rowBase + r) * K + k];
  __syncthreads();
  int hi8 = (lane & 16) ? 8 : 0;
  int c0  = lane & 15;
  v8f acc[4];
  for (int j = 0; j < 4; ++j) {
    if (flags & 2) {
      for (int r = 0; r < 8; ++r)
        acc[j][r] = out[(size_t)(rowBase + hi8 + r) * N + colBase + j * 16 + c0];
    } else {
      for (int r = 0; r < 8; ++r) acc[j][r] = 0.f;
    }
  }
  int ntilesTot = N >> 4;
  for (int kt = 0; kt < (K >> 5); ++kt) {
    v16h a = a_frag(ldsA, K, kt, lane);
    for (int j = 0; j < 4; ++j) {
      v16h b = b_frag(Wpk, kt * ntilesTot + (colBase >> 4) + j, lane);
      acc[j] = WMMA_F16(a, b, acc[j]);
    }
  }
  for (int j = 0; j < 4; ++j) {
    int col = colBase + j * 16 + c0;
    float bv = bias ? bias[col] : 0.f;
    for (int r = 0; r < 8; ++r) {
      float v = acc[j][r] + bv;
      if (flags & 1) v = siluf(v);
      out[(size_t)(rowBase + hi8 + r) * N + col] = v;
    }
  }
}

// ---- Fused pairwise message block (the hot path) ----
// 1 wave handles (b, n, 16 m's). mode: 0=init edge embedding, 1=layer (logits+w), 2=final layer (logits+edge upd)
__global__ __launch_bounds__(32) void pair_msg(
    const float* __restrict__ eq,        // [B,N,3,64] equi (or equi_n)
    const float* __restrict__ sA,        // [B*N,64]  inv @ W_s
    const float* __restrict__ rA,        // [B*N,64]  inv @ W_r
    const float* __restrict__ b_m,       // [64]
    const _Float16* __restrict__ Wd_pk,  // 2kt x 4nt
    const _Float16* __restrict__ We_pk,  // ktE x 4nt
    int ktE,
    const _Float16* __restrict__ Wf1_pk, // 2kt x 8nt
    const float* __restrict__ bf1,       // [128]
    const _Float16* __restrict__ Wf2_pk, // 4kt x 8nt
    const float* __restrict__ bf2,       // [128]
    const int* __restrict__ bond_types,  // mode0
    const float* __restrict__ bond_tab,  // [5,64] mode0
    const _Float16* __restrict__ edges_in,  // [B,N,N,128] f16, modes 1/2
    _Float16* __restrict__ edges_out,       // mode0: ff ; mode2: edges + ff@W_edge + b_edge
    const _Float16* __restrict__ Wedge_pk,  // mode2: 4kt x 8nt
    const float* __restrict__ b_edge,       // mode2
    const float* __restrict__ W_attn,       // [128,8]
    const float* __restrict__ b_attn,       // [8]
    float* __restrict__ logits,             // [B,N,N,8]
    const float* __restrict__ W_cw,         // [128] mode1 (may be null)
    float* __restrict__ wout,               // [B,N,N]
    int mode) {
  __shared__ __align__(16) _Float16 sD[16 * 64];
  __shared__ __align__(16) _Float16 sE[16 * 128];
  __shared__ __align__(16) _Float16 sM[16 * 64];
  __shared__ __align__(16) _Float16 sT[16 * 128];
  __shared__ float sF[16 * 128];
  int lane = threadIdx.x;
  int mt = blockIdx.x, n = blockIdx.y, b = blockIdx.z;
  int m0 = mt * 16;
  // warm the packed weight streams while VALU computes the dots
  __builtin_prefetch(Wf1_pk + (size_t)lane * 16, 0, 0);
  __builtin_prefetch(Wf2_pk + (size_t)lane * 16, 0, 0);
  // edge input: async DMA the 16x128 f16 tile straight into LDS (ASYNCcnt path)
  if (mode != 0) {
    const _Float16* ein = edges_in + (((size_t)(b * NATOM + n)) * NATOM + m0) * 128;
    unsigned ldsBase = lds_off(sE);
    for (int it = 0; it < 8; ++it) {
      int e8 = (it * 32 + lane) * 8;  // halves
      async_b128(ldsBase + e8 * 2, ein + e8);
    }
  }
  const float* eqrow = eq + (size_t)(b * NATOM + n) * 192;
  // dots[m',d] = sum_i eq[n,i,d]*eq[m,i,d]
  for (int idx = lane; idx < 16 * 64; idx += 32) {
    int mr = idx >> 6, d = idx & 63;
    const float* eqm = eq + (size_t)(b * NATOM + m0 + mr) * 192;
    float s = eqrow[d] * eqm[d] + eqrow[64 + d] * eqm[64 + d] + eqrow[128 + d] * eqm[128 + d];
    sD[mr * 64 + d] = (_Float16)s;
  }
  if (mode == 0) {
    for (int idx = lane; idx < 16 * 64; idx += 32) {
      int mr = idx >> 6, d = idx & 63;
      int bt = bond_types[((size_t)(b * NATOM + n)) * NATOM + m0 + mr];
      sE[mr * 128 + d] = (_Float16)bond_tab[bt * 64 + d];
    }
  } else {
    async_wait();
  }
  __syncthreads();
  int hi8 = (lane & 16) ? 8 : 0;
  int c0  = lane & 15;
  // m = silu(s_n + r_m + dots@W_d + e@W_e + b_m)
  v8f macc[4];
  for (int j = 0; j < 4; ++j) for (int r = 0; r < 8; ++r) macc[j][r] = 0.f;
  for (int kt = 0; kt < 2; ++kt) {
    v16h a = a_frag(sD, 64, kt, lane);
    for (int j = 0; j < 4; ++j) macc[j] = WMMA_F16(a, b_frag(Wd_pk, kt * 4 + j, lane), macc[j]);
  }
  for (int kt = 0; kt < ktE; ++kt) {
    v16h a = a_frag(sE, 128, kt, lane);
    for (int j = 0; j < 4; ++j) macc[j] = WMMA_F16(a, b_frag(We_pk, kt * 4 + j, lane), macc[j]);
  }
  const float* sv = sA + (size_t)(b * NATOM + n) * 64;
  for (int j = 0; j < 4; ++j) {
    int col = j * 16 + c0;
    for (int r = 0; r < 8; ++r) {
      int row = hi8 + r;
      float v = macc[j][r] + sv[col] + rA[(size_t)(b * NATOM + m0 + row) * 64 + col] + b_m[col];
      sM[row * 64 + col] = (_Float16)siluf(v);
    }
  }
  __syncthreads();
  // t = silu(m @ Wf1 + bf1)
  v8f tacc[8];
  for (int j = 0; j < 8; ++j) for (int r = 0; r < 8; ++r) tacc[j][r] = 0.f;
  for (int kt = 0; kt < 2; ++kt) {
    v16h a = a_frag(sM, 64, kt, lane);
    for (int j = 0; j < 8; ++j) tacc[j] = WMMA_F16(a, b_frag(Wf1_pk, kt * 8 + j, lane), tacc[j]);
  }
  for (int j = 0; j < 8; ++j) {
    int col = j * 16 + c0;
    for (int r = 0; r < 8; ++r) sT[(hi8 + r) * 128 + col] = (_Float16)siluf(tacc[j][r] + bf1[col]);
  }
  __syncthreads();
  // ff = t @ Wf2 + bf2
  v8f facc[8];
  for (int j = 0; j < 8; ++j) for (int r = 0; r < 8; ++r) facc[j][r] = 0.f;
  for (int kt = 0; kt < 4; ++kt) {
    v16h a = a_frag(sT, 128, kt, lane);
    for (int j = 0; j < 8; ++j) facc[j] = WMMA_F16(a, b_frag(Wf2_pk, kt * 8 + j, lane), facc[j]);
  }
  if (mode == 0) {
    _Float16* eo = edges_out + (((size_t)(b * NATOM + n)) * NATOM + m0) * 128;
    for (int j = 0; j < 8; ++j) {
      int col = j * 16 + c0;
      for (int r = 0; r < 8; ++r) eo[(hi8 + r) * 128 + col] = (_Float16)(facc[j][r] + bf2[col]);
    }
    return;
  }
  for (int j = 0; j < 8; ++j) {
    int col = j * 16 + c0;
    for (int r = 0; r < 8; ++r) sF[(hi8 + r) * 128 + col] = facc[j][r] + bf2[col];
  }
  __syncthreads();
  // attention logits and coord weights from ff
  for (int idx = lane; idx < 16 * 8; idx += 32) {
    int row = idx >> 3, h = idx & 7;
    float acc = b_attn[h];
    for (int c = 0; c < 128; ++c) acc += sF[row * 128 + c] * W_attn[c * 8 + h];
    logits[((((size_t)(b * NATOM + n)) * NATOM) + m0 + row) * 8 + h] = acc;
  }
  if (mode == 1 && W_cw) {
    for (int row = lane; row < 16; row += 32) {
      float acc = 0.f;
      for (int c = 0; c < 128; ++c) acc += sF[row * 128 + c] * W_cw[c];
      wout[((size_t)(b * NATOM + n)) * NATOM + m0 + row] = acc;
    }
  }
  if (mode == 2) {
    __syncthreads();
    for (int idx = lane; idx < 16 * 128; idx += 32) sT[idx] = (_Float16)sF[idx];
    __syncthreads();
    v8f eacc[8];
    const _Float16* ein = edges_in + (((size_t)(b * NATOM + n)) * NATOM + m0) * 128;
    for (int j = 0; j < 8; ++j) {
      int col = j * 16 + c0;
      for (int r = 0; r < 8; ++r) eacc[j][r] = (float)ein[(hi8 + r) * 128 + col] + b_edge[col];
    }
    for (int kt = 0; kt < 4; ++kt) {
      v16h a = a_frag(sT, 128, kt, lane);
      for (int j = 0; j < 8; ++j) eacc[j] = WMMA_F16(a, b_frag(Wedge_pk, kt * 8 + j, lane), eacc[j]);
    }
    _Float16* eo = edges_out + (((size_t)(b * NATOM + n)) * NATOM + m0) * 128;
    for (int j = 0; j < 8; ++j) {
      int col = j * 16 + c0;
      for (int r = 0; r < 8; ++r) eo[(hi8 + r) * 128 + col] = (_Float16)eacc[j][r];
    }
  }
}

// ---- BondRefine: m=silu(s_n+r_m+dist*Wd+e@We+b); t=silu(m@Wf1+bf1)[256]; er=e+t@Wf2+bf2 ----
__global__ __launch_bounds__(32) void bond_refine(
    const float* __restrict__ coordsOut, const float* __restrict__ sA, const float* __restrict__ rA,
    const float* __restrict__ Wd_row, const float* __restrict__ b_m,
    const _Float16* __restrict__ We_pk, const _Float16* __restrict__ Wf1_pk,
    const float* __restrict__ bf1, const _Float16* __restrict__ Wf2_pk, const float* __restrict__ bf2,
    const _Float16* __restrict__ edges_in, _Float16* __restrict__ edges_ref) {
  __shared__ __align__(16) _Float16 sE[16 * 128];
  __shared__ __align__(16) _Float16 sM[16 * 64];
  __shared__ __align__(16) _Float16 sT[16 * 256];
  __shared__ float sDist[16];
  int lane = threadIdx.x;
  int mt = blockIdx.x, n = blockIdx.y, b = blockIdx.z;
  int m0 = mt * 16;
  __builtin_prefetch(Wf1_pk + (size_t)lane * 16, 0, 0);
  __builtin_prefetch(Wf2_pk + (size_t)lane * 16, 0, 0);
  const _Float16* ein = edges_in + (((size_t)(b * NATOM + n)) * NATOM + m0) * 128;
  {
    unsigned ldsBase = lds_off(sE);
    for (int it = 0; it < 8; ++it) {
      int e8 = (it * 32 + lane) * 8;
      async_b128(ldsBase + e8 * 2, ein + e8);
    }
  }
  const float* cn = coordsOut + (size_t)(b * NATOM + n) * 3;
  for (int row = lane; row < 16; row += 32) {
    const float* cm = coordsOut + (size_t)(b * NATOM + m0 + row) * 3;
    float dx = cn[0] - cm[0], dy = cn[1] - cm[1], dz = cn[2] - cm[2];
    sDist[row] = sqrtf(dx * dx + dy * dy + dz * dz + 1e-6f);
  }
  async_wait();
  __syncthreads();
  int hi8 = (lane & 16) ? 8 : 0;
  int c0  = lane & 15;
  v8f macc[4];
  for (int j = 0; j < 4; ++j) for (int r = 0; r < 8; ++r) macc[j][r] = 0.f;
  for (int kt = 0; kt < 4; ++kt) {
    v16h a = a_frag(sE, 128, kt, lane);
    for (int j = 0; j < 4; ++j) macc[j] = WMMA_F16(a, b_frag(We_pk, kt * 4 + j, lane), macc[j]);
  }
  const float* sv = sA + (size_t)(b * NATOM + n) * 64;
  for (int j = 0; j < 4; ++j) {
    int col = j * 16 + c0;
    for (int r = 0; r < 8; ++r) {
      int row = hi8 + r;
      float v = macc[j][r] + sv[col] + rA[(size_t)(b * NATOM + m0 + row) * 64 + col]
              + sDist[row] * Wd_row[col] + b_m[col];
      sM[row * 64 + col] = (_Float16)siluf(v);
    }
  }
  __syncthreads();
  v8f tacc[16];
  for (int j = 0; j < 16; ++j) for (int r = 0; r < 8; ++r) tacc[j][r] = 0.f;
  for (int kt = 0; kt < 2; ++kt) {
    v16h a = a_frag(sM, 64, kt, lane);
    for (int j = 0; j < 16; ++j) tacc[j] = WMMA_F16(a, b_frag(Wf1_pk, kt * 16 + j, lane), tacc[j]);
  }
  for (int j = 0; j < 16; ++j) {
    int col = j * 16 + c0;
    for (int r = 0; r < 8; ++r) sT[(hi8 + r) * 256 + col] = (_Float16)siluf(tacc[j][r] + bf1[col]);
  }
  __syncthreads();
  v8f facc[8];
  for (int j = 0; j < 8; ++j) {
    int col = j * 16 + c0;
    for (int r = 0; r < 8; ++r) facc[j][r] = (float)sE[(hi8 + r) * 128 + col] + bf2[col];
  }
  for (int kt = 0; kt < 8; ++kt) {
    v16h a = a_frag(sT, 256, kt, lane);
    for (int j = 0; j < 8; ++j) facc[j] = WMMA_F16(a, b_frag(Wf2_pk, kt * 8 + j, lane), facc[j]);
  }
  _Float16* eo = edges_ref + (((size_t)(b * NATOM + n)) * NATOM + m0) * 128;
  for (int j = 0; j < 8; ++j) {
    int col = j * 16 + c0;
    for (int r = 0; r < 8; ++r) eo[(hi8 + r) * 128 + col] = (_Float16)facc[j][r];
  }
}

// ---- small VALU / reduction kernels ----
__global__ void ln_kernel(const float* __restrict__ x, const float* __restrict__ g,
                          const float* __restrict__ bta, float* __restrict__ out, int D) {
  __shared__ float red[256];
  int t = threadIdx.x;
  size_t row = blockIdx.x;
  float v = x[row * D + t];
  red[t] = v; __syncthreads();
  for (int s = 128; s > 0; s >>= 1) { if (t < s) red[t] += red[t + s]; __syncthreads(); }
  float mean = red[0] / (float)D; __syncthreads();
  float c = v - mean;
  red[t] = c * c; __syncthreads();
  for (int s = 128; s > 0; s >>= 1) { if (t < s) red[t] += red[t + s]; __syncthreads(); }
  float rstd = rsqrtf(red[0] / (float)D + 1e-5f);
  out[row * D + t] = c * rstd * g[t] + bta[t];
}

__global__ void coord_norm_kernel(const float* __restrict__ equi, const float* __restrict__ mask,
                                  const float* __restrict__ gamma, float* __restrict__ out) {
  int d = threadIdx.x, b = blockIdx.x;
  float suml = 0.f, summ = 0.f;
  for (int n = 0; n < NATOM; ++n) {
    const float* e = equi + (size_t)(b * NATOM + n) * 192;
    float l = sqrtf(e[d] * e[d] + e[64 + d] * e[64 + d] + e[128 + d] * e[128 + d] + 1e-6f);
    float mk = mask[b * NATOM + n];
    suml += l * mk; summ += mk;
  }
  float sc = gamma[d] / (suml / summ + 1e-6f);
  for (int n = 0; n < NATOM; ++n) {
    const float* e = equi + (size_t)(b * NATOM + n) * 192;
    float* o = out + (size_t)(b * NATOM + n) * 192;
    o[d] = e[d] * sc; o[64 + d] = e[64 + d] * sc; o[128 + d] = e[128 + d] * sc;
  }
}

__global__ void feats_kernel(const float* __restrict__ times, const int* __restrict__ at,
                             const int* __restrict__ ch, const float* __restrict__ atab,
                             const float* __restrict__ ctab, float* __restrict__ feats) {
  int t = threadIdx.x;
  size_t row = (size_t)(blockIdx.y * NATOM + blockIdx.x);
  float v;
  if (t < 64) v = atab[at[row] * 64 + t];
  else if (t < 128) v = ctab[ch[row] * 64 + (t - 64)];
  else {
    int j = t - 128, jj = j & 31;
    float freq = __expf((float)jj * (logf(1000.f) / 31.f));
    float ang = times[row] * freq;
    v = (j < 32) ? __sinf(ang) : __cosf(ang);
  }
  feats[row * 192 + t] = v;
}

__global__ void equi_embed_kernel(const float* __restrict__ coords, const float* __restrict__ times,
                                  const float* __restrict__ Wce, const float* __restrict__ Wt1,
                                  const float* __restrict__ bt1, const float* __restrict__ Wt2,
                                  const float* __restrict__ bt2, float* __restrict__ equi) {
  __shared__ float h[64];
  int d = threadIdx.x;
  size_t row = (size_t)(blockIdx.y * NATOM + blockIdx.x);
  float tm = times[row];
  h[d] = siluf(tm * Wt1[d] + bt1[d]);
  __syncthreads();
  float te = bt2[d];
  for (int j = 0; j < 64; ++j) te += h[j] * Wt2[j * 64 + d];
  float sc = Wce[d] * (1.f + te);
  for (int i = 0; i < 3; ++i) equi[row * 192 + i * 64 + d] = coords[row * 3 + i] * sc;
}

__global__ void softmax_kernel(float* __restrict__ logits, const float* __restrict__ mask,
                               float* __restrict__ meanout) {
  int n = blockIdx.x, b = blockIdx.y, t = threadIdx.x;
  float* L = logits + ((size_t)(b * NATOM + n)) * NATOM * 8;
  float mkn = mask[b * NATOM + n];
  if (t < 8) {
    float mx = -1e30f;
    for (int m = 0; m < NATOM; ++m) {
      float pm = mkn * mask[b * NATOM + m] * ((m == n) ? 0.f : 1.f);
      float v = (pm > 0.f) ? L[m * 8 + t] : -1e9f;
      L[m * 8 + t] = v;
      mx = fmaxf(mx, v);
    }
    float sum = 0.f;
    for (int m = 0; m < NATOM; ++m) { float e = __expf(L[m * 8 + t] - mx); L[m * 8 + t] = e; sum += e; }
    for (int m = 0; m < NATOM; ++m) {
      float pm = mkn * mask[b * NATOM + m] * ((m == n) ? 0.f : 1.f);
      L[m * 8 + t] = L[m * 8 + t] / sum * pm;
    }
  }
  __syncthreads();
  for (int m = t; m < NATOM; m += 32) {
    float s = 0.f;
    for (int h = 0; h < 8; ++h) s += L[m * 8 + h];
    meanout[((size_t)(b * NATOM + n)) * NATOM + m] = s * 0.125f;
  }
}

__global__ void inv_attn_kernel(const float* __restrict__ attn, const float* __restrict__ vals,
                                float* __restrict__ out) {
  int n = blockIdx.x, b = blockIdx.y, t = threadIdx.x;
  int h = t >> 5, d = t & 31;
  const float* A = attn + ((size_t)(b * NATOM + n)) * NATOM * 8;
  float acc = 0.f;
  for (int m = 0; m < NATOM; ++m) acc += A[m * 8 + h] * vals[(size_t)(b * NATOM + m) * 256 + h * 32 + d];
  out[(size_t)(b * NATOM + n) * 256 + t] = acc;
}

__global__ void equi_attn_kernel(const float* __restrict__ mean, const float* __restrict__ eqv,
                                 float* __restrict__ out) {
  int n = blockIdx.x, b = blockIdx.y, t = threadIdx.x;  // t = i*64+d, 192 threads
  const float* Mn = mean + ((size_t)(b * NATOM + n)) * NATOM;
  float acc = 0.f;
  for (int m = 0; m < NATOM; ++m) acc += Mn[m] * eqv[(size_t)(b * NATOM + m) * 192 + t];
  out[(size_t)(b * NATOM + n) * 192 + t] = acc;
}

__global__ void coord_update_kernel(const float* __restrict__ cin, float* __restrict__ cout,
                                    const float* __restrict__ w, const float* __restrict__ mask) {
  int n = blockIdx.x, b = blockIdx.y, t = threadIdx.x;
  float mkn = mask[b * NATOM + n];
  float den = 1.f;
  for (int m = 0; m < NATOM; ++m) den += mkn * mask[b * NATOM + m] * ((m == n) ? 0.f : 1.f);
  if (t < 3) {
    float xn = cin[(size_t)(b * NATOM + n) * 3 + t];
    float s = 0.f;
    for (int m = 0; m < NATOM; ++m) {
      float pm = mkn * mask[b * NATOM + m] * ((m == n) ? 0.f : 1.f);
      s += w[((size_t)(b * NATOM + n)) * NATOM + m] * (xn - cin[(size_t)(b * NATOM + m) * 3 + t]) * pm;
    }
    cout[(size_t)(b * NATOM + n) * 3 + t] = xn + s / den;
  }
}

__global__ void coords_out_kernel(const float* __restrict__ eqf, const float* __restrict__ Wco,
                                  const float* __restrict__ coords, float* __restrict__ out) {
  int n = blockIdx.x, b = blockIdx.y, t = threadIdx.x;
  if (t < 3) {
    const float* e = eqf + (size_t)(b * NATOM + n) * 192 + t * 64;
    float s = 0.f;
    for (int d = 0; d < 64; ++d) s += e[d] * Wco[d];
    out[(size_t)(b * NATOM + n) * 3 + t] = s + coords[(size_t)(b * NATOM + n) * 3 + t];
  }
}

__global__ void heads_kernel(const float* __restrict__ invf, const float* __restrict__ Wa,
                             const float* __restrict__ ba, const float* __restrict__ Wc,
                             const float* __restrict__ bc, float* __restrict__ outA,
                             float* __restrict__ outC) {
  int n = blockIdx.x, b = blockIdx.y, t = threadIdx.x;
  const float* x = invf + (size_t)(b * NATOM + n) * 256;
  if (t < 15) {
    float s = ba[t];
    for (int k = 0; k < 256; ++k) s += x[k] * Wa[k * 15 + t];
    outA[(size_t)(b * NATOM + n) * 15 + t] = s;
  }
  if (t < 8) {
    float s = bc[t];
    for (int k = 0; k < 256; ++k) s += x[k] * Wc[k * 8 + t];
    outC[(size_t)(b * NATOM + n) * 8 + t] = s;
  }
}

__global__ void bond_logits_kernel(const _Float16* __restrict__ er, const float* __restrict__ g,
                                   const float* __restrict__ be_, const float* __restrict__ Wb,
                                   const float* __restrict__ bb, float* __restrict__ bl) {
  __shared__ float red[32];
  __shared__ float xn[128];
  int m = blockIdx.x, n = blockIdx.y, b = blockIdx.z, t = threadIdx.x;
  const _Float16* e = er + (((size_t)(b * NATOM + n)) * NATOM + m) * 128;
  float s = 0.f;
  for (int k = t; k < 128; k += 32) s += (float)e[k];
  red[t] = s; __syncthreads();
  for (int st = 16; st > 0; st >>= 1) { if (t < st) red[t] += red[t + st]; __syncthreads(); }
  float mean = red[0] / 128.f; __syncthreads();
  s = 0.f;
  for (int k = t; k < 128; k += 32) { float c = (float)e[k] - mean; s += c * c; }
  red[t] = s; __syncthreads();
  for (int st = 16; st > 0; st >>= 1) { if (t < st) red[t] += red[t + st]; __syncthreads(); }
  float rstd = rsqrtf(red[0] / 128.f + 1e-5f);
  for (int k = t; k < 128; k += 32) xn[k] = ((float)e[k] - mean) * rstd * g[k] + be_[k];
  __syncthreads();
  if (t < 5) {
    float acc = bb[t];
    for (int k = 0; k < 128; ++k) acc += xn[k] * Wb[k * 5 + t];
    bl[(((size_t)(b * NATOM + n)) * NATOM + m) * 5 + t] = acc;
  }
}

__global__ void symmetrize_kernel(const float* __restrict__ bl, float* __restrict__ out) {
  size_t idx = (size_t)blockIdx.x * 256 + threadIdx.x;
  size_t total = (size_t)BATCH * NATOM * NATOM * 5;
  if (idx >= total) return;
  int c = idx % 5;
  size_t r = idx / 5;
  int m = r % NATOM;
  size_t r2 = r / NATOM;
  int n = r2 % NATOM;
  int b = r2 / NATOM;
  float a  = bl[(((size_t)(b * NATOM + n)) * NATOM + m) * 5 + c];
  float bs = bl[(((size_t)(b * NATOM + m)) * NATOM + n) * 5 + c];
  out[idx] = 0.5f * (a + bs);
}

// ========================= host orchestration =========================
struct LayerP {
  const float *g1, *b1, *gcn, *Ws, *Wr, *Wd, *We, *bm, *Wf1, *bf1, *Wf2, *bf2,
      *Wattn, *battn, *Wval, *Wout, *bout, *Wev, *Weo, *g2, *b2, *Wff1, *bff1, *Wff2, *bff2,
      *Wcw, *Wedge, *bedge;
};

extern "C" void kernel_launch(void* const* d_in, const int* in_sizes, int n_in,
                              void* d_out, int out_size, void* d_ws, size_t ws_size,
                              hipStream_t stream) {
  const float* in_coords = (const float*)d_in[0];
  const float* in_times  = (const float*)d_in[1];
  const int*   in_atom   = (const int*)d_in[2];
  const int*   in_charge = (const int*)d_in[3];
  const int*   in_bond   = (const int*)d_in[4];
  const float* in_mask   = (const float*)d_in[5];

  int pi = 6;
  auto P = [&]() { return (const float*)d_in[pi++]; };
  const float* atom_tab   = P();
  const float* charge_tab = P();
  const float* W_inv_emb  = P();
  const float* b_inv_emb  = P();
  const float* bond_tab   = P();
  const float *beWs = P(), *beWr = P(), *beWd = P(), *beWe = P(), *bebm = P(),
              *beWf1 = P(), *bebf1 = P(), *beWf2 = P(), *bebf2 = P();
  const float* W_coord_emb = P();
  const float *Wt1 = P(), *bt1 = P(), *Wt2 = P(), *bt2 = P();
  LayerP Lp[6];
  for (int i = 0; i < 6; ++i) {
    LayerP& L = Lp[i];
    L.g1 = P(); L.b1 = P(); L.gcn = P(); L.Ws = P(); L.Wr = P(); L.Wd = P(); L.We = P();
    L.bm = P(); L.Wf1 = P(); L.bf1 = P(); L.Wf2 = P(); L.bf2 = P(); L.Wattn = P(); L.battn = P();
    L.Wval = P(); L.Wout = P(); L.bout = P(); L.Wev = P(); L.Weo = P(); L.g2 = P(); L.b2 = P();
    L.Wff1 = P(); L.bff1 = P(); L.Wff2 = P(); L.bff2 = P();
    if (i < 5) { L.Wcw = P(); L.Wedge = nullptr; L.bedge = nullptr; }
    else       { L.Wcw = nullptr; L.Wedge = P(); L.bedge = P(); }
  }
  const float* gamma_cn_f = P();
  const float* W_coord_out = P();
  const float *g_inv_f = P(), *b_inv_f = P();
  const float *W_atom = P(), *b_atom = P(), *W_charge = P(), *b_charge = P();
  const float *brWs = P(), *brWr = P(), *brWd = P(), *brWe = P(), *brbm = P(),
              *brWf1 = P(), *brbf1 = P(), *brWf2 = P(), *brbf2 = P();
  const float *g_bond = P(), *b_bond = P(), *W_bond = P(), *b_bond_p = P();

  // ---- workspace carve ----
  char* cur = (char*)d_ws;
  auto allocf = [&](size_t n) { float* p = (float*)cur; cur += ((n * 4 + 255) / 256) * 256; return p; };
  auto alloch = [&](size_t n) { _Float16* p = (_Float16*)cur; cur += ((n * 2 + 255) / 256) * 256; return p; };
  float* coordsA = allocf((size_t)BN * 3);
  float* coordsB = allocf((size_t)BN * 3);
  float* feats   = allocf((size_t)BN * 192);
  float* inv     = allocf((size_t)BN * 256);
  float* invn    = allocf((size_t)BN * 256);
  float* equi    = allocf((size_t)BN * 192);
  float* eqn     = allocf((size_t)BN * 192);
  float* sbuf    = allocf((size_t)BN * 64);
  float* rbuf    = allocf((size_t)BN * 64);
  float* vals    = allocf((size_t)BN * 256);
  float* eqv     = allocf((size_t)BN * 192);
  float* invAtt  = allocf((size_t)BN * 256);
  float* eqAtt   = allocf((size_t)BN * 192);
  float* hbuf    = allocf((size_t)BN * 256);
  float* ff1     = allocf((size_t)BN * 512);
  float* logits  = allocf((size_t)BN * NATOM * 8);
  float* meanat  = allocf((size_t)BN * NATOM);
  float* wbuf    = allocf((size_t)BN * NATOM);
  float* bl      = allocf((size_t)BN * NATOM * 5);
  _Float16* edges    = alloch((size_t)BN * NATOM * 128);
  _Float16* edgesRef = alloch((size_t)BN * NATOM * 128);

  auto pack = [&](const float* W, int K, int Ncol) {
    int frags = (K / 32) * (Ncol / 16);
    _Float16* dst = alloch((size_t)frags * 512);
    pack_weight<<<dim3(frags), dim3(32), 0, stream>>>(W, dst, K, Ncol);
    return (const _Float16*)dst;
  };
  auto gemm = [&](const float* A, const _Float16* Wpk, const float* bias, float* out,
                  int M, int K, int Ncol, int flags) {
    gemm_wmma<<<dim3(M / 16, Ncol / 64), dim3(32), 16 * K * 2, stream>>>(A, Wpk, bias, out, M, K, Ncol, flags);
  };

  const _Float16* pkEmb  = pack(W_inv_emb, 192, 256);
  const _Float16* pkBeWs = pack(beWs, 256, 64);
  const _Float16* pkBeWr = pack(beWr, 256, 64);
  const _Float16* pkBeWd = pack(beWd, 64, 64);
  const _Float16* pkBeWe = pack(beWe, 64, 64);
  const _Float16* pkBeWf1 = pack(beWf1, 64, 128);
  const _Float16* pkBeWf2 = pack(beWf2, 128, 128);
  const _Float16 *pkWs[6], *pkWr[6], *pkWd[6], *pkWe[6], *pkWf1[6], *pkWf2[6],
                 *pkWval[6], *pkWout[6], *pkWev[6], *pkWeo[6], *pkWff1[6], *pkWff2[6];
  const _Float16* pkWedge = nullptr;
  for (int i = 0; i < 6; ++i) {
    pkWs[i] = pack(Lp[i].Ws, 256, 64);   pkWr[i] = pack(Lp[i].Wr, 256, 64);
    pkWd[i] = pack(Lp[i].Wd, 64, 64);    pkWe[i] = pack(Lp[i].We, 128, 64);
    pkWf1[i] = pack(Lp[i].Wf1, 64, 128); pkWf2[i] = pack(Lp[i].Wf2, 128, 128);
    pkWval[i] = pack(Lp[i].Wval, 256, 256); pkWout[i] = pack(Lp[i].Wout, 256, 256);
    pkWev[i] = pack(Lp[i].Wev, 64, 64);  pkWeo[i] = pack(Lp[i].Weo, 64, 64);
    pkWff1[i] = pack(Lp[i].Wff1, 256, 512); pkWff2[i] = pack(Lp[i].Wff2, 512, 256);
    if (i == 5) pkWedge = pack(Lp[i].Wedge, 128, 128);
  }
  const _Float16* pkBrWs  = pack(brWs, 256, 64);
  const _Float16* pkBrWr  = pack(brWr, 256, 64);
  const _Float16* pkBrWe  = pack(brWe, 128, 64);
  const _Float16* pkBrWf1 = pack(brWf1, 64, 256);
  const _Float16* pkBrWf2 = pack(brWf2, 256, 128);

  float* outCoords = (float*)d_out;
  float* outAtom   = outCoords + (size_t)BN * 3;
  float* outCharge = outAtom + (size_t)BN * 15;
  float* outBond   = outCharge + (size_t)BN * 8;

  hipMemcpyAsync(coordsA, in_coords, (size_t)BN * 3 * sizeof(float), hipMemcpyDeviceToDevice, stream);

  // ---- embeddings ----
  feats_kernel<<<dim3(NATOM, BATCH), dim3(192), 0, stream>>>(in_times, in_atom, in_charge, atom_tab, charge_tab, feats);
  gemm(feats, pkEmb, b_inv_emb, inv, BN, 192, 256, 1);
  equi_embed_kernel<<<dim3(NATOM, BATCH), dim3(64), 0, stream>>>(in_coords, in_times, W_coord_emb, Wt1, bt1, Wt2, bt2, equi);

  // ---- edge embedding (pair block, mode 0) ----
  gemm(inv, pkBeWs, nullptr, sbuf, BN, 256, 64, 0);
  gemm(inv, pkBeWr, nullptr, rbuf, BN, 256, 64, 0);
  pair_msg<<<dim3(NATOM / 16, NATOM, BATCH), dim3(32), 0, stream>>>(
      equi, sbuf, rbuf, bebm, pkBeWd, pkBeWe, 2, pkBeWf1, bebf1, pkBeWf2, bebf2,
      in_bond, bond_tab, nullptr, edges, nullptr, nullptr,
      nullptr, nullptr, nullptr, nullptr, nullptr, 0);

  float* cOld = coordsA;
  float* cNew = coordsB;
  for (int i = 0; i < 6; ++i) {
    const LayerP& L = Lp[i];
    ln_kernel<<<dim3(BN), dim3(256), 0, stream>>>(inv, L.g1, L.b1, invn, 256);
    coord_norm_kernel<<<dim3(BATCH), dim3(64), 0, stream>>>(equi, in_mask, L.gcn, eqn);
    gemm(invn, pkWs[i], nullptr, sbuf, BN, 256, 64, 0);
    gemm(invn, pkWr[i], nullptr, rbuf, BN, 256, 64, 0);
    gemm(invn, pkWval[i], nullptr, vals, BN, 256, 256, 0);
    gemm(eqn, pkWev[i], nullptr, eqv, BN * 3, 64, 64, 0);
    int mode = (i == 5) ? 2 : 1;
    pair_msg<<<dim3(NATOM / 16, NATOM, BATCH), dim3(32), 0, stream>>>(
        eqn, sbuf, rbuf, L.bm, pkWd[i], pkWe[i], 4, pkWf1[i], L.bf1, pkWf2[i], L.bf2,
        nullptr, nullptr, edges, edges, (i == 5) ? pkWedge : nullptr, L.bedge,
        L.Wattn, L.battn, logits, L.Wcw, wbuf, mode);
    softmax_kernel<<<dim3(NATOM, BATCH), dim3(32), 0, stream>>>(logits, in_mask, meanat);
    inv_attn_kernel<<<dim3(NATOM, BATCH), dim3(256), 0, stream>>>(logits, vals, invAtt);
    gemm(invAtt, pkWout[i], L.bout, inv, BN, 256, 256, 2);
    equi_attn_kernel<<<dim3(NATOM, BATCH), dim3(192), 0, stream>>>(meanat, eqv, eqAtt);
    gemm(eqAtt, pkWeo[i], nullptr, equi, BN * 3, 64, 64, 2);
    ln_kernel<<<dim3(BN), dim3(256), 0, stream>>>(inv, L.g2, L.b2, hbuf, 256);
    gemm(hbuf, pkWff1[i], L.bff1, ff1, BN, 256, 512, 1);
    gemm(ff1, pkWff2[i], L.bff2, inv, BN, 512, 256, 2);
    if (i < 5) {
      coord_update_kernel<<<dim3(NATOM, BATCH), dim3(32), 0, stream>>>(cOld, cNew, wbuf, in_mask);
      float* tmp = cOld; cOld = cNew; cNew = tmp;
    }
  }

  // ---- output heads ----
  coord_norm_kernel<<<dim3(BATCH), dim3(64), 0, stream>>>(equi, in_mask, gamma_cn_f, eqn);
  coords_out_kernel<<<dim3(NATOM, BATCH), dim3(32), 0, stream>>>(eqn, W_coord_out, cOld, outCoords);
  ln_kernel<<<dim3(BN), dim3(256), 0, stream>>>(inv, g_inv_f, b_inv_f, invn, 256);
  heads_kernel<<<dim3(NATOM, BATCH), dim3(32), 0, stream>>>(invn, W_atom, b_atom, W_charge, b_charge, outAtom, outCharge);

  // ---- BondRefine + bond logits ----
  gemm(invn, pkBrWs, nullptr, sbuf, BN, 256, 64, 0);
  gemm(invn, pkBrWr, nullptr, rbuf, BN, 256, 64, 0);
  bond_refine<<<dim3(NATOM / 16, NATOM, BATCH), dim3(32), 0, stream>>>(
      outCoords, sbuf, rbuf, brWd, brbm, pkBrWe, pkBrWf1, brbf1, pkBrWf2, brbf2, edges, edgesRef);
  bond_logits_kernel<<<dim3(NATOM, NATOM, BATCH), dim3(32), 0, stream>>>(edgesRef, g_bond, b_bond, W_bond, b_bond_p, bl);
  size_t totalBond = (size_t)BATCH * NATOM * NATOM * 5;
  symmetrize_kernel<<<dim3((unsigned)((totalBond + 255) / 256)), dim3(256), 0, stream>>>(bl, outBond);
}
